// Llama4MoE_78993038508065
// MI455X (gfx1250) — compile-verified, matrix operand order checked by reading, not run on hardware
//
#include <hip/hip_runtime.h>
#include <hip/hip_bf16.h>
#include <stdint.h>

// ---- problem constants (match reference) ----
#define T_TOK 1024
#define H_DIM 2048
#define I_DIM 4096
#define N_EXP 8

// ---- tiling ----
#define BM 64
#define BN 64   // output columns per block (per weight matrix)
#define BK 32   // one WMMA K-step
#define WST 40  // LDS row stride (ushorts) for transposed W tile; 80B rows -> 16B aligned
#define MAX_RT ((T_TOK + N_EXP * (BM - 1)) / BM)  // 23 padded row-tiles worst case
#define MAX_ROWS (MAX_RT * BM)                    // 1472

typedef __attribute__((ext_vector_type(16))) __bf16 v16bf;
typedef __attribute__((ext_vector_type(8))) float v8f;

union FragBF {
  v16bf v;
  uint4 q[2];
};

__device__ __forceinline__ unsigned short f2bf(float f) {
  unsigned int u = __float_as_uint(f);
  u += 0x7FFFu + ((u >> 16) & 1u);  // round-to-nearest-even
  return (unsigned short)(u >> 16);
}

__device__ __forceinline__ unsigned int pack2bf(float a, float b) {
  return (unsigned int)f2bf(a) | ((unsigned int)f2bf(b) << 16);
}

// ---------------------------------------------------------------------------
// Router: logits = x @ w_router, top-1 (first-max tie-break), sigmoid scale.
// ---------------------------------------------------------------------------
__global__ void router_kernel(const float* __restrict__ x,
                              const float* __restrict__ wr,
                              int* __restrict__ expertIdx,
                              float* __restrict__ scale) {
  int t = blockIdx.x * blockDim.x + threadIdx.x;
  if (t >= T_TOK) return;
  float acc[N_EXP];
#pragma unroll
  for (int e = 0; e < N_EXP; ++e) acc[e] = 0.f;
  const float* xr = x + (size_t)t * H_DIM;
  for (int k = 0; k < H_DIM; ++k) {
    float xv = xr[k];
    const float* w = wr + (size_t)k * N_EXP;
#pragma unroll
    for (int e = 0; e < N_EXP; ++e) acc[e] += xv * w[e];
  }
  int best = 0;
  float bv = acc[0];
#pragma unroll
  for (int e = 1; e < N_EXP; ++e)
    if (acc[e] > bv) { bv = acc[e]; best = e; }
  expertIdx[t] = best;
  scale[t] = 1.f / (1.f + __expf(-bv));
}

// ---------------------------------------------------------------------------
// Build deterministic BM-padded permutation grouped by expert.
// ---------------------------------------------------------------------------
__global__ void build_perm_kernel(const int* __restrict__ expertIdx,
                                  int* __restrict__ row_to_token,
                                  int* __restrict__ tileExpert,
                                  int* __restrict__ nTiles) {
  __shared__ int cnt[N_EXP];
  __shared__ int tileOff[N_EXP + 1];
  int tid = threadIdx.x;
  if (tid < N_EXP) {
    int c = 0;
    for (int t = 0; t < T_TOK; ++t) c += (expertIdx[t] == tid);
    cnt[tid] = c;
  }
  __syncthreads();
  if (tid == 0) {
    int to = 0;
    for (int e = 0; e < N_EXP; ++e) {
      tileOff[e] = to;
      int nt = (cnt[e] + BM - 1) / BM;
      for (int j = 0; j < nt; ++j) tileExpert[to + j] = e;
      to += nt;
    }
    tileOff[N_EXP] = to;
    *nTiles = to;
  }
  __syncthreads();
  if (tid < N_EXP) {
    int base = tileOff[tid] * BM;
    int n = 0;
    for (int t = 0; t < T_TOK; ++t)
      if (expertIdx[t] == tid) row_to_token[base + n++] = t;
    int cap = ((cnt[tid] + BM - 1) / BM) * BM;
    for (; n < cap; ++n) row_to_token[base + n] = -1;
  }
}

// ---------------------------------------------------------------------------
// GEMM1: h = silu(x@Wg) * (x@Wu).  64x64 block tile per weight, 8 waves
// (4 row x 2 col), each wave: 16 rows x 32 cols -> 2 accumulators per weight.
// fp32 weights converted to bf16 in the LDS fill (packed b128 stores).
// ---------------------------------------------------------------------------
template <bool ROUTED>
__global__ void gemm1_kernel(const float* __restrict__ x,
                             const float* __restrict__ scale,
                             const int* __restrict__ row_to_token,
                             const int* __restrict__ tileExpert,
                             const int* __restrict__ nTiles,
                             const float* __restrict__ wg_base,
                             const float* __restrict__ wu_base,
                             unsigned short* __restrict__ h_out) {
  __shared__ alignas(16) unsigned short sX[BM * BK];    // 4 KB
  __shared__ alignas(16) unsigned short sWg[BN * WST];  // 5 KB (transposed [n][k])
  __shared__ alignas(16) unsigned short sWu[BN * WST];  // 5 KB

  int rt = blockIdx.x;
  const float* wg = wg_base;
  const float* wu = wu_base;
  if (ROUTED) {
    if (rt >= *nTiles) return;  // block-uniform: EXEC stays all-1s for WMMA
    size_t off = (size_t)tileExpert[rt] * H_DIM * I_DIM;
    wg += off;
    wu += off;
  }
  int r0 = rt * BM;
  int nBase = blockIdx.y * BN;

  // ---- x-tile fill coords (1 thread = 8 k-elems of one row) ----
  int fr = threadIdx.x >> 2;
  int fk = (threadIdx.x & 3) * 8;
  int ftok;
  float fs;
  if (ROUTED) {
    ftok = row_to_token[r0 + fr];
    fs = (ftok >= 0) ? scale[ftok] : 0.f;  // pad rows: scale 0 -> zero tile, no branch
    ftok = (ftok >= 0) ? ftok : 0;
  } else {
    ftok = r0 + fr;
    fs = 1.f;
  }
  const float* xrow = x + (size_t)ftok * H_DIM + fk;

  // ---- weight-tile fill coords (1 thread = 8 consecutive k of one n) ----
  int wn = threadIdx.x & 63;
  int wk0 = (threadIdx.x >> 6) * 8;

  // ---- wave / WMMA coords ----
  int lane = threadIdx.x & 31;
  int wave = threadIdx.x >> 5;
  int mw = wave >> 1, nw = wave & 1;
  int half = lane >> 4, ln = lane & 15;

  v8f accg[2] = {};
  v8f accu[2] = {};

  for (int kb = 0; kb < H_DIM; kb += BK) {
    {  // x tile: scale + convert + pack -> one ds_store_b128
      const float4 x0 = *reinterpret_cast<const float4*>(xrow + kb);
      const float4 x1 = *reinterpret_cast<const float4*>(xrow + kb + 4);
      uint4 p;
      p.x = pack2bf(x0.x * fs, x0.y * fs);
      p.y = pack2bf(x0.z * fs, x0.w * fs);
      p.z = pack2bf(x1.x * fs, x1.y * fs);
      p.w = pack2bf(x1.z * fs, x1.w * fs);
      *reinterpret_cast<uint4*>(&sX[fr * BK + fk]) = p;
    }
    {  // weight tiles: 8 k-strided b32 reads (coalesced over n), packed b128 store
      const float* gp = wg + (size_t)(kb + wk0) * I_DIM + nBase + wn;
      const float* up = wu + (size_t)(kb + wk0) * I_DIM + nBase + wn;
      float gv[8], uv[8];
#pragma unroll
      for (int j = 0; j < 8; ++j) {
        gv[j] = gp[(size_t)j * I_DIM];
        uv[j] = up[(size_t)j * I_DIM];
      }
      uint4 pg, pu;
      pg.x = pack2bf(gv[0], gv[1]); pg.y = pack2bf(gv[2], gv[3]);
      pg.z = pack2bf(gv[4], gv[5]); pg.w = pack2bf(gv[6], gv[7]);
      pu.x = pack2bf(uv[0], uv[1]); pu.y = pack2bf(uv[2], uv[3]);
      pu.z = pack2bf(uv[4], uv[5]); pu.w = pack2bf(uv[6], uv[7]);
      *reinterpret_cast<uint4*>(&sWg[wn * WST + wk0]) = pg;
      *reinterpret_cast<uint4*>(&sWu[wn * WST + wk0]) = pu;
      if (kb + BK < H_DIM) {  // prefetch next K-panel of the streamed operand
        __builtin_prefetch(gp + (size_t)BK * I_DIM, 0, 0);
        __builtin_prefetch(up + (size_t)BK * I_DIM, 0, 0);
      }
    }
    __syncthreads();

    FragBF a;
    // 16-bit A layout (ISA 7.12.2): lanes 0-15 K=0..7/16..23, lanes 16-31 K=8..15/24..31
    a.q[0] = *reinterpret_cast<const uint4*>(&sX[(16 * mw + ln) * BK + half * 8]);
    a.q[1] = *reinterpret_cast<const uint4*>(&sX[(16 * mw + ln) * BK + 16 + half * 8]);
#pragma unroll
    for (int j = 0; j < 2; ++j) {
      int nrow = (32 * nw + 16 * j + ln) * WST;
      FragBF bg, bu;
      bg.q[0] = *reinterpret_cast<const uint4*>(&sWg[nrow + half * 8]);
      bg.q[1] = *reinterpret_cast<const uint4*>(&sWg[nrow + 16 + half * 8]);
      bu.q[0] = *reinterpret_cast<const uint4*>(&sWu[nrow + half * 8]);
      bu.q[1] = *reinterpret_cast<const uint4*>(&sWu[nrow + 16 + half * 8]);
      accg[j] = __builtin_amdgcn_wmma_f32_16x16x32_bf16(false, a.v, false, bg.v,
                                                        (short)0, accg[j], false, false);
      accu[j] = __builtin_amdgcn_wmma_f32_16x16x32_bf16(false, a.v, false, bu.v,
                                                        (short)0, accu[j], false, false);
    }
    __syncthreads();
  }

  // epilogue: silu(g)*u -> bf16 h   (D layout: VGPR i -> row i + 8*half, col ln)
#pragma unroll
  for (int j = 0; j < 2; ++j) {
#pragma unroll
    for (int i = 0; i < 8; ++i) {
      float g = accg[j][i], u = accu[j][i];
      float hv = (g / (1.f + __expf(-g))) * u;
      int r = r0 + 16 * mw + 8 * half + i;
      int c = nBase + 32 * nw + 16 * j + ln;
      h_out[(size_t)r * I_DIM + c] = f2bf(hv);
    }
  }
}

// ---------------------------------------------------------------------------
// GEMM2: out = h @ Wd.  Shared path writes '='; routed path scatter-adds '+='.
// ---------------------------------------------------------------------------
template <bool ROUTED>
__global__ void gemm2_kernel(const unsigned short* __restrict__ h,
                             const float* __restrict__ wd_base,
                             const int* __restrict__ row_to_token,
                             const int* __restrict__ tileExpert,
                             const int* __restrict__ nTiles,
                             float* __restrict__ out) {
  __shared__ alignas(16) unsigned short sH[BM * BK];
  __shared__ alignas(16) unsigned short sW[BN * WST];

  int rt = blockIdx.x;
  const float* wd = wd_base;
  if (ROUTED) {
    if (rt >= *nTiles) return;
    wd += (size_t)tileExpert[rt] * I_DIM * H_DIM;
  }
  int r0 = rt * BM;
  int nBase = blockIdx.y * BN;

  int fr = threadIdx.x >> 2;
  int fk = (threadIdx.x & 3) * 8;
  int wn = threadIdx.x & 63;
  int wk0 = (threadIdx.x >> 6) * 8;

  int lane = threadIdx.x & 31;
  int wave = threadIdx.x >> 5;
  int mw = wave >> 1, nw = wave & 1;
  int half = lane >> 4, ln = lane & 15;

  v8f acc[2] = {};
  const unsigned short* hrow = h + (size_t)(r0 + fr) * I_DIM + fk;

  for (int kb = 0; kb < I_DIM; kb += BK) {
    *reinterpret_cast<uint4*>(&sH[fr * BK + fk]) =
        *reinterpret_cast<const uint4*>(hrow + kb);  // bf16 passthrough, 16B
    {
      const float* wp = wd + (size_t)(kb + wk0) * H_DIM + nBase + wn;
      float wv[8];
#pragma unroll
      for (int j = 0; j < 8; ++j) wv[j] = wp[(size_t)j * H_DIM];
      uint4 p;
      p.x = pack2bf(wv[0], wv[1]); p.y = pack2bf(wv[2], wv[3]);
      p.z = pack2bf(wv[4], wv[5]); p.w = pack2bf(wv[6], wv[7]);
      *reinterpret_cast<uint4*>(&sW[wn * WST + wk0]) = p;
      if (kb + BK < I_DIM) __builtin_prefetch(wp + (size_t)BK * H_DIM, 0, 0);
    }
    __syncthreads();

    FragBF a;
    a.q[0] = *reinterpret_cast<const uint4*>(&sH[(16 * mw + ln) * BK + half * 8]);
    a.q[1] = *reinterpret_cast<const uint4*>(&sH[(16 * mw + ln) * BK + 16 + half * 8]);
#pragma unroll
    for (int j = 0; j < 2; ++j) {
      int nrow = (32 * nw + 16 * j + ln) * WST;
      FragBF b;
      b.q[0] = *reinterpret_cast<const uint4*>(&sW[nrow + half * 8]);
      b.q[1] = *reinterpret_cast<const uint4*>(&sW[nrow + 16 + half * 8]);
      acc[j] = __builtin_amdgcn_wmma_f32_16x16x32_bf16(false, a.v, false, b.v,
                                                       (short)0, acc[j], false, false);
    }
    __syncthreads();
  }

#pragma unroll
  for (int j = 0; j < 2; ++j) {
#pragma unroll
    for (int i = 0; i < 8; ++i) {
      int r = r0 + 16 * mw + 8 * half + i;
      int c = nBase + 32 * nw + 16 * j + ln;
      if (ROUTED) {
        int tok = row_to_token[r];
        if (tok >= 0) out[(size_t)tok * H_DIM + c] += acc[j][i];  // disjoint rows/cols
      } else {
        out[(size_t)r * H_DIM + c] = acc[j][i];
      }
    }
  }
}

// ---------------------------------------------------------------------------
extern "C" void kernel_launch(void* const* d_in, const int* in_sizes, int n_in,
                              void* d_out, int out_size, void* d_ws, size_t ws_size,
                              hipStream_t stream) {
  (void)in_sizes; (void)n_in; (void)out_size; (void)ws_size;
  const float* x        = (const float*)d_in[0];
  const float* w_router = (const float*)d_in[1];
  const float* ws_gate  = (const float*)d_in[2];
  const float* ws_up    = (const float*)d_in[3];
  const float* ws_down  = (const float*)d_in[4];
  const float* we_gate  = (const float*)d_in[5];
  const float* we_up    = (const float*)d_in[6];
  const float* we_down  = (const float*)d_in[7];
  float* out = (float*)d_out;

  char* ws = (char*)d_ws;
  int*   expertIdx    = (int*)(ws + 0);       // 4 KB
  float* scale        = (float*)(ws + 4096);  // 4 KB
  int*   row_to_token = (int*)(ws + 8192);    // 6 KB (MAX_ROWS)
  int*   tileExpert   = (int*)(ws + 14336);   // 128 B
  int*   nTiles       = (int*)(ws + 14464);   // 4 B
  unsigned short* h_shared = (unsigned short*)(ws + 32768);                              // 8 MB
  unsigned short* h_routed = (unsigned short*)(ws + 32768 + (size_t)T_TOK * I_DIM * 2);  // 11.5 MB

  router_kernel<<<dim3((T_TOK + 255) / 256), dim3(256), 0, stream>>>(
      x, w_router, expertIdx, scale);
  build_perm_kernel<<<dim3(1), dim3(32), 0, stream>>>(
      expertIdx, row_to_token, tileExpert, nTiles);

  gemm1_kernel<false><<<dim3(T_TOK / BM, I_DIM / BN), dim3(256), 0, stream>>>(
      x, scale, nullptr, nullptr, nullptr, ws_gate, ws_up, h_shared);
  gemm1_kernel<true><<<dim3(MAX_RT, I_DIM / BN), dim3(256), 0, stream>>>(
      x, scale, row_to_token, tileExpert, nTiles, we_gate, we_up, h_routed);

  gemm2_kernel<false><<<dim3(T_TOK / BM, H_DIM / BN), dim3(256), 0, stream>>>(
      h_shared, ws_down, nullptr, nullptr, nullptr, out);
  gemm2_kernel<true><<<dim3(MAX_RT, H_DIM / BN), dim3(256), 0, stream>>>(
      h_routed, we_down, row_to_token, tileExpert, nTiles, out);
}
